// DGMG_62208306315835
// MI455X (gfx1250) — compile-verified, matrix operand order checked by reading, not run on hardware
//
#include <hip/hip_runtime.h>
#include <hip/hip_bf16.h>

// ---------------------------------------------------------------------------
// DGMG forward loss on MI455X (gfx1250).
// GEMMs run on v_wmma_f32_16x16x32_f16 (f16 in / f32 accumulate), with
// double-buffered K-tiles (loads overlap WMMA), lane-major packed B fragments
// (2x global_load_b128 per fragment), and v_cvt_pk_rtz_f16_f32 A conversion.
// Readout uses the identity segment_sum(hV@gpW + gpb) == (sum_n hV)@gpW + N*gpb.
// ---------------------------------------------------------------------------

#define Bg   256    // graphs
#define Nn   128    // nodes per graph
#define Dd   256    // node hidden
#define Gg   512    // graph hidden
#define Ee   (Bg*Nn*8)
#define Ss   4
#define Tt   2
#define CH   4096   // fs row chunk (32 graphs)

typedef __attribute__((ext_vector_type(16))) _Float16 v16h;
typedef __attribute__((ext_vector_type(2)))  __fp16   fp16x2;   // cvt_pkrtz result type
typedef __attribute__((ext_vector_type(8)))  float    v8f;

union FragA { fp16x2 h2[8]; v16h v; };
union FragB { uint4  q[2];  v16h v; };
union HU    { fp16x2 h;     unsigned u; };

static __device__ __forceinline__ float sigmoidf_(float x) {
    return 1.0f / (1.0f + __expf(-x));
}

// --------------------------- utility kernels -------------------------------

__global__ void k_zero_f32(float* __restrict__ p, long n) {
    long i = (long)blockIdx.x * blockDim.x + threadIdx.x;
    if (i < n) p[i] = 0.0f;
}

// Pack W[K,N] (f32 row-major) into lane-major f16 WMMA B-fragment order.
// Fragment (kt,jt) = 256 uints; uint index = lane*8+v; lane: n=jt*16+(lane&15),
// kb=(lane<16?0:8); dword v holds (K0,K0+1), K0 = kt*32+((v>>2)<<4)+kb+2*(v&3).
__global__ void k_pack_f16(const float* __restrict__ W, unsigned* __restrict__ out,
                           int K, int N) {
    int tid = blockIdx.x * blockDim.x + threadIdx.x;
    int total = (K * N) >> 1;
    if (tid >= total) return;
    int frag = tid >> 8;
    int r    = tid & 255;
    int lane = r >> 3;
    int v    = r & 7;
    int NT = N >> 4;
    int kt = frag / NT;
    int jt = frag - kt * NT;
    int n  = (jt << 4) + (lane & 15);
    int kb = (lane < 16) ? 0 : 8;
    int K0 = (kt << 5) + ((v >> 2) << 4) + kb + ((v & 3) << 1);
    HU h;
    h.h = __builtin_amdgcn_cvt_pkrtz(W[(long)K0 * N + n], W[(long)(K0 + 1) * N + n]);
    out[tid] = h.u;
}

// ------------------------------ WMMA GEMM ----------------------------------
// C[M,N] = act(A[M,K] @ Bpack + biasScale*bias), A f32 (cvt_pk_rtz to f16 on
// the fly), Bpack pre-packed f16 fragments, C f32.
// ACT: 0=none 1=sigmoid 2=relu.  M%16==0, K%64==0, N%64==0.
// One wave computes a 16x64 strip; K-tiles double-buffered.

static __device__ __forceinline__ void loadA(FragA& a, const float* __restrict__ Arow,
                                             int kt, int kb) {
#pragma unroll
    for (int g = 0; g < 2; ++g) {
        const float4* p = (const float4*)(Arow + (kt << 5) + (g << 4) + kb);
        float4 f0 = p[0];
        float4 f1 = p[1];
        a.h2[g * 4 + 0] = __builtin_amdgcn_cvt_pkrtz(f0.x, f0.y);
        a.h2[g * 4 + 1] = __builtin_amdgcn_cvt_pkrtz(f0.z, f0.w);
        a.h2[g * 4 + 2] = __builtin_amdgcn_cvt_pkrtz(f1.x, f1.y);
        a.h2[g * 4 + 3] = __builtin_amdgcn_cvt_pkrtz(f1.z, f1.w);
    }
}

static __device__ __forceinline__ void loadB4(FragB* b, const uint4* __restrict__ Bp,
                                              size_t fragBase, int lane) {
    const uint4* p = Bp + fragBase + (lane << 1);
#pragma unroll
    for (int j = 0; j < 4; ++j) {
        b[j].q[0] = p[(j << 6) + 0];   // 64 uint4 per fragment
        b[j].q[1] = p[(j << 6) + 1];
    }
}

template <int ACT>
__global__ void __launch_bounds__(256)
k_gemm_f16(const float* __restrict__ A, const uint4* __restrict__ Bp,
           const float* __restrict__ bias, float biasScale,
           float* __restrict__ C, int M, int K, int N) {
    const int lane   = threadIdx.x & 31;
    const int wid    = (blockIdx.x * blockDim.x + threadIdx.x) >> 5;
    const int nWaves = (gridDim.x * blockDim.x) >> 5;
    const int NT64   = N >> 6;
    const int nStrips = (M >> 4) * NT64;
    const int NT = N >> 4;
    const int KT = K >> 5;           // even for all K used (256/512/768)
    const int kb = (lane < 16) ? 0 : 8;

    for (int s = wid; s < nStrips; s += nWaves) {
        const int mt  = s / NT64;
        const int jt0 = (s - mt * NT64) << 2;
        v8f acc[4] = {v8f{}, v8f{}, v8f{}, v8f{}};
        const int m = (mt << 4) + (lane & 15);
        const float* Arow = A + (size_t)m * K;
        FragA a0, a1;
        FragB b0[4], b1[4];
        loadA(a0, Arow, 0, kb);
        loadB4(b0, Bp, ((size_t)jt0) << 6, lane);
        for (int kt = 0; kt < KT; kt += 2) {
            // prefetch odd tile while even tile computes
            loadA(a1, Arow, kt + 1, kb);
            loadB4(b1, Bp, ((size_t)((kt + 1) * NT + jt0)) << 6, lane);
#pragma unroll
            for (int j = 0; j < 4; ++j)
                acc[j] = __builtin_amdgcn_wmma_f32_16x16x32_f16(
                    false, a0.v, false, b0[j].v, (short)0, acc[j], false, false);
            if (kt + 2 < KT) {
                loadA(a0, Arow, kt + 2, kb);
                loadB4(b0, Bp, ((size_t)((kt + 2) * NT + jt0)) << 6, lane);
            }
#pragma unroll
            for (int j = 0; j < 4; ++j)
                acc[j] = __builtin_amdgcn_wmma_f32_16x16x32_f16(
                    false, a1.v, false, b1[j].v, (short)0, acc[j], false, false);
        }
        // Epilogue.  C layout: VGPR i -> row mt*16 + i + 8*(lane>=16), col = lane&15.
        const int rbase = (mt << 4) + ((lane >> 4) << 3);
        const int colb  = (jt0 << 4) + (lane & 15);
#pragma unroll
        for (int j = 0; j < 4; ++j) {
            int col = colb + (j << 4);
            float bv = bias[col] * biasScale;
#pragma unroll
            for (int i = 0; i < 8; ++i) {
                float val = acc[j][i] + bv;
                if (ACT == 1)      val = sigmoidf_(val);
                else if (ACT == 2) val = fmaxf(val, 0.0f);
                C[(size_t)(rbase + i) * N + col] = val;
            }
        }
    }
}

// ----------------------------- graph kernels -------------------------------

// nodeSum[b,d] = sum_n hV[b*Nn+n, d]
__global__ void k_node_sum(const float* __restrict__ hV, float* __restrict__ ns) {
    int tid = blockIdx.x * blockDim.x + threadIdx.x;   // B*D threads
    if (tid >= Bg * Dd) return;
    int d = tid & (Dd - 1);
    int b = tid >> 8;
    const float* base = hV + (size_t)b * Nn * Dd + d;
    float s = 0.0f;
    for (int n = 0; n < Nn; ++n) s += base[(size_t)n * Dd];
    ns[tid] = s;
}

__global__ void k_scatter_hv(float* __restrict__ hV, const float* __restrict__ hv,
                             int pos) {
    int tid = blockIdx.x * blockDim.x + threadIdx.x;   // B*D
    if (tid >= Bg * Dd) return;
    int d = tid & (Dd - 1);
    int b = tid >> 8;
    hV[((size_t)b * Nn + pos) * Dd + d] = hv[(size_t)b * Dd + d];
}

// edge message: m[dst] += hV[src]  (4 floats / thread, coalesced per edge)
__global__ void k_edge_scatter(const float* __restrict__ hV,
                               const int* __restrict__ src,
                               const int* __restrict__ dst,
                               float* __restrict__ m) {
    long tid = (long)blockIdx.x * blockDim.x + threadIdx.x;  // E * D/4
    if (tid >= (long)Ee * (Dd >> 2)) return;
    int e = (int)(tid >> 6);            // D/4 == 64
    int d = ((int)tid & 63) << 2;
    const float* s = hV + (size_t)src[e] * Dd + d;
    float*       p = m  + (size_t)dst[e] * Dd + d;
    atomicAdd(p + 0, s[0]);
    atomicAdd(p + 1, s[1]);
    atomicAdd(p + 2, s[2]);
    atomicAdd(p + 3, s[3]);
}

// fan loss: logits = fanH @ w2 + b2 (2 classes), -mean(log_softmax[lab])
__global__ void k_fan_loss(const float* __restrict__ fanH,
                           const float* __restrict__ w2, const float* __restrict__ b2,
                           const int* __restrict__ lab, float* __restrict__ loss) {
    int b = threadIdx.x;   // 256 threads, 1 block
    float x0 = b2[0], x1 = b2[1];
    const float* h = fanH + (size_t)b * Gg;
    for (int i = 0; i < Gg; ++i) { x0 += h[i] * w2[2 * i]; x1 += h[i] * w2[2 * i + 1]; }
    float mx  = fmaxf(x0, x1);
    float lse = mx + __logf(__expf(x0 - mx) + __expf(x1 - mx));
    float lp  = ((lab[b] == 0) ? x0 : x1) - lse;
    atomicAdd(loss, -lp * (1.0f / Bg));
}

__global__ void k_concat_fae(const float* __restrict__ hG, const float* __restrict__ hv,
                             float* __restrict__ X) {
    int tid = blockIdx.x * blockDim.x + threadIdx.x;   // B*(G+D)
    if (tid >= Bg * (Gg + Dd)) return;
    int b = tid / (Gg + Dd);
    int c = tid - b * (Gg + Dd);
    X[tid] = (c < Gg) ? hG[(size_t)b * Gg + c] : hv[(size_t)b * Dd + (c - Gg)];
}

// fae loss: pe = sigmoid(faeH @ w2 + b2); binary CE vs lab_e
__global__ void k_fae_loss(const float* __restrict__ faeH,
                           const float* __restrict__ w2, const float* __restrict__ b2,
                           const int* __restrict__ lab, float* __restrict__ loss) {
    int b = threadIdx.x;   // 256 threads, 1 block
    float x = b2[0];
    const float* h = faeH + (size_t)b * (Gg + Dd);
    for (int i = 0; i < Gg + Dd; ++i) x += h[i] * w2[i];
    float pe = sigmoidf_(x);
    float l  = (float)lab[b];
    float v  = l * __logf(pe + 1e-7f) + (1.0f - l) * __logf(1.0f - pe + 1e-7f);
    atomicAdd(loss, -v * (1.0f / Bg));
}

__global__ void k_concat_fs(const float* __restrict__ hV, const float* __restrict__ hv,
                            float* __restrict__ X, int rowStart) {
    int tid = blockIdx.x * blockDim.x + threadIdx.x;   // CH * 2D
    if (tid >= CH * 2 * Dd) return;
    int i = tid >> 9;                 // 2D == 512
    int c = tid & 511;
    int r = rowStart + i;
    int b = r >> 7;                   // Nn == 128
    X[tid] = (c < Dd) ? hV[(size_t)r * Dd + c] : hv[(size_t)b * Dd + (c - Dd)];
}

__global__ void k_fs_score(const float* __restrict__ H, const float* __restrict__ w2,
                           const float* __restrict__ b2, float* __restrict__ scores,
                           int rowStart) {
    int i = blockIdx.x * blockDim.x + threadIdx.x;     // CH rows
    if (i >= CH) return;
    const float* h = H + (size_t)i * (2 * Dd);
    float x = b2[0];
    for (int k = 0; k < 2 * Dd; ++k) x += h[k] * w2[k];
    scores[rowStart + i] = x;
}

__global__ void k_fs_loss(const float* __restrict__ scores, const int* __restrict__ sel,
                          float* __restrict__ loss) {
    int b = threadIdx.x;   // 256 threads, 1 block
    const float* s = scores + (size_t)b * Nn;
    float mx = -3.4e38f;
    for (int n = 0; n < Nn; ++n) mx = fmaxf(mx, s[n]);
    float sum = 0.0f;
    for (int n = 0; n < Nn; ++n) sum += __expf(s[n] - mx);
    float lp = s[sel[b]] - (mx + __logf(sum));
    atomicAdd(loss, -lp * (1.0f / Bg));
}

// ------------------------------- host side ---------------------------------

static inline void gemm(const float* A, const unsigned* Bp, const float* bias,
                        float biasScale, float* C, int M, int K, int N, int act,
                        hipStream_t st) {
    int strips = (M >> 4) * (N >> 6);
    int blocks = (strips + 7) / 8;
    if (blocks < 1) blocks = 1;
    const uint4* B4 = (const uint4*)Bp;
    if (act == 0)
        k_gemm_f16<0><<<blocks, 256, 0, st>>>(A, B4, bias, biasScale, C, M, K, N);
    else if (act == 1)
        k_gemm_f16<1><<<blocks, 256, 0, st>>>(A, B4, bias, biasScale, C, M, K, N);
    else
        k_gemm_f16<2><<<blocks, 256, 0, st>>>(A, B4, bias, biasScale, C, M, K, N);
}

extern "C" void kernel_launch(void* const* d_in, const int* in_sizes, int n_in,
                              void* d_out, int out_size, void* d_ws, size_t ws_size,
                              hipStream_t stream) {
    // inputs (setup_inputs order)
    const float* hV0      = (const float*)d_in[0];
    const float* gcn_W    = (const float*)d_in[1];   // [T,D,D]
    const float* gcn_b    = (const float*)d_in[2];   // [T,D]
    const float* gpW      = (const float*)d_in[3];   // [D,G]
    const float* gpb      = (const float*)d_in[4];
    const float* fan_w1   = (const float*)d_in[5];
    const float* fan_b1   = (const float*)d_in[6];
    const float* fan_w2   = (const float*)d_in[7];
    const float* fan_b2   = (const float*)d_in[8];
    const float* fae_w1   = (const float*)d_in[9];
    const float* fae_b1   = (const float*)d_in[10];
    const float* fae_w2   = (const float*)d_in[11];
    const float* fae_b2   = (const float*)d_in[12];
    const float* fs_w1    = (const float*)d_in[13];
    const float* fs_b1    = (const float*)d_in[14];
    const float* fs_w2    = (const float*)d_in[15];
    const float* fs_b2    = (const float*)d_in[16];
    const float* finit_w1 = (const float*)d_in[17];
    const float* finit_b1 = (const float*)d_in[18];
    const float* finit_w2 = (const float*)d_in[19];
    const float* finit_b2 = (const float*)d_in[20];
    const int*   src      = (const int*)d_in[21];
    const int*   dst      = (const int*)d_in[22];
    const int*   lab_n    = (const int*)d_in[23];    // [S,B]
    const int*   lab_e    = (const int*)d_in[24];
    const int*   nsel     = (const int*)d_in[25];
    float* loss = (float*)d_out;

    // workspace carve-out
    char* base = (char*)d_ws;
    size_t cur = 0;
    auto allocF = [&](size_t n) { float* p = (float*)(base + cur);
                                  cur += ((n * 4 + 255) & ~(size_t)255); return p; };
    auto allocU = [&](size_t n) { unsigned* p = (unsigned*)(base + cur);
                                  cur += ((n * 4 + 255) & ~(size_t)255); return p; };
    float* hV      = allocF((size_t)Bg * Nn * Dd);       // 32 MB
    float* msg     = allocF((size_t)Bg * Nn * Dd);       // 32 MB
    float* Xc      = allocF((size_t)CH * 2 * Dd);        // 8 MB
    float* Hc      = allocF((size_t)CH * 2 * Dd);        // 8 MB
    float* hG      = allocF((size_t)Bg * Gg);
    float* nodeSum = allocF((size_t)Bg * Dd);
    float* fanH    = allocF((size_t)Bg * Gg);
    float* finitH  = allocF((size_t)Bg * Gg);
    float* hv      = allocF((size_t)Bg * Dd);
    float* faeX    = allocF((size_t)Bg * (Gg + Dd));
    float* faeH    = allocF((size_t)Bg * (Gg + Dd));
    float* scores  = allocF((size_t)Bg * Nn);
    unsigned* p_gcn0  = allocU((size_t)Dd * Dd / 2);
    unsigned* p_gcn1  = allocU((size_t)Dd * Dd / 2);
    unsigned* p_gpW   = allocU((size_t)Dd * Gg / 2);
    unsigned* p_fan1  = allocU((size_t)Gg * Gg / 2);
    unsigned* p_fae1  = allocU((size_t)(Gg + Dd) * (Gg + Dd) / 2);
    unsigned* p_fs1   = allocU((size_t)(2 * Dd) * (2 * Dd) / 2);
    unsigned* p_fi1   = allocU((size_t)Gg * Gg / 2);
    unsigned* p_fi2   = allocU((size_t)Gg * Dd / 2);
    (void)ws_size; (void)in_sizes; (void)n_in; (void)out_size;

    // ---- prep: loss=0, hV=hV0, pack weights to f16 fragment layout ----
    k_zero_f32<<<1, 32, 0, stream>>>(loss, 1);
    (void)hipMemcpyAsync(hV, hV0, (size_t)Bg * Nn * Dd * 4,
                         hipMemcpyDeviceToDevice, stream);
    auto pack = [&](const float* W, unsigned* out, int K, int N) {
        int tot = (K * N) >> 1;
        k_pack_f16<<<(tot + 255) / 256, 256, 0, stream>>>(W, out, K, N);
    };
    pack(gcn_W,                p_gcn0, Dd, Dd);
    pack(gcn_W + Dd * Dd,      p_gcn1, Dd, Dd);
    pack(gpW,                  p_gpW,  Dd, Gg);
    pack(fan_w1,               p_fan1, Gg, Gg);
    pack(fae_w1,               p_fae1, Gg + Dd, Gg + Dd);
    pack(fs_w1,                p_fs1,  2 * Dd, 2 * Dd);
    pack(finit_w1,             p_fi1,  Gg, Gg);
    pack(finit_w2,             p_fi2,  Gg, Dd);

    const int BD = Bg * Dd;
    // initial readout: hG = nodeSum @ gpW + N*gpb
    k_node_sum<<<(BD + 255) / 256, 256, 0, stream>>>(hV, nodeSum);
    gemm(nodeSum, p_gpW, gpb, (float)Nn, hG, Bg, Dd, Gg, 0, stream);

    for (int s = 0; s < Ss; ++s) {
        int pos = s % Nn;
        // --- node step: fan (add-node decision) ---
        gemm(hG, p_fan1, fan_b1, 1.0f, fanH, Bg, Gg, Gg, 1, stream);
        k_fan_loss<<<1, 256, 0, stream>>>(fanH, fan_w2, fan_b2, lab_n + s * Bg, loss);
        // --- finit: new node repr, scatter, readout ---
        gemm(hG, p_fi1, finit_b1, 1.0f, finitH, Bg, Gg, Gg, 1, stream);
        gemm(finitH, p_fi2, finit_b2, 1.0f, hv, Bg, Gg, Dd, 0, stream);
        k_scatter_hv<<<(BD + 255) / 256, 256, 0, stream>>>(hV, hv, pos);
        k_node_sum<<<(BD + 255) / 256, 256, 0, stream>>>(hV, nodeSum);
        gemm(nodeSum, p_gpW, gpb, (float)Nn, hG, Bg, Dd, Gg, 0, stream);
        // --- edge step: fae (add-edge decision) ---
        int nfae = Bg * (Gg + Dd);
        k_concat_fae<<<(nfae + 255) / 256, 256, 0, stream>>>(hG, hv, faeX);
        gemm(faeX, p_fae1, fae_b1, 1.0f, faeH, Bg, Gg + Dd, Gg + Dd, 1, stream);
        k_fae_loss<<<1, 256, 0, stream>>>(faeH, fae_w2, fae_b2, lab_e + s * Bg, loss);
        // --- fs: score all B*N candidates against the new node (chunked) ---
        for (int r0 = 0; r0 < Bg * Nn; r0 += CH) {
            int nc = CH * 2 * Dd;
            k_concat_fs<<<(nc + 255) / 256, 256, 0, stream>>>(hV, hv, Xc, r0);
            gemm(Xc, p_fs1, fs_b1, 1.0f, Hc, CH, 2 * Dd, 2 * Dd, 1, stream);
            k_fs_score<<<(CH + 255) / 256, 256, 0, stream>>>(Hc, fs_w2, fs_b2,
                                                             scores, r0);
        }
        k_fs_loss<<<1, 256, 0, stream>>>(scores, nsel + s * Bg, loss);
        // --- propagate: T GCN layers ---
        for (int t = 0; t < Tt; ++t) {
            long nm = (long)Bg * Nn * Dd;
            k_zero_f32<<<(int)((nm + 255) / 256), 256, 0, stream>>>(msg, nm);
            long ne = (long)Ee * (Dd >> 2);
            k_edge_scatter<<<(int)((ne + 255) / 256), 256, 0, stream>>>(hV, src, dst, msg);
            gemm(msg, (t == 0) ? p_gcn0 : p_gcn1, gcn_b + t * Dd, 1.0f,
                 hV, Bg * Nn, Dd, Dd, 2, stream);
        }
        // --- readout after propagate ---
        k_node_sum<<<(BD + 255) / 256, 256, 0, stream>>>(hV, nodeSum);
        gemm(nodeSum, p_gpW, gpb, (float)Nn, hG, Bg, Dd, Gg, 0, stream);
    }
}
